// Attn_19688130085636
// MI455X (gfx1250) — matrix operand order repaired and checked by reading, not compiled, not run on hardware
//
#include <hip/hip_runtime.h>

typedef __attribute__((ext_vector_type(16))) __bf16 v16bf;
typedef __attribute__((ext_vector_type(8)))  float  v8f;
typedef int v4i __attribute__((vector_size(16)));   // matches builtin param type

#define HIDDEN_DIM 1024
#define SEQ_LEN    8192
#define STATE_LEN  4096

// CDNA5 async global->LDS staging path (gfx1250): gate on builtin presence so
// we gracefully fall back to register-staged ds_store if the toolchain lacks it.
#if __has_builtin(__builtin_amdgcn_global_load_async_to_lds_b128) && \
    __has_builtin(__builtin_amdgcn_s_wait_asynccnt)
#define USE_ASYNC_LDS 1
#else
#define USE_ASYNC_LDS 0
#endif

// Exact parameter types per hipcc diagnostic: global side is
// 'v4i __device__*' (addrspace(1)); LDS side is addrspace(3).
typedef __attribute__((address_space(1))) v4i* gv4i_ptr;
typedef __attribute__((address_space(3))) v4i* lv4i_ptr;
#define GPTR128(x) ((gv4i_ptr)(void*)(x))
#define LPTR128(x) ((lv4i_ptr)(void*)(x))

// ---------------------------------------------------------------------------
// f32 -> bf16 bulk convert, 4 elements/thread (global_load_b128 + b64 store)
// ---------------------------------------------------------------------------
__global__ __launch_bounds__(256)
void f32_to_bf16_kernel(const float* __restrict__ src,
                        __bf16* __restrict__ dst, int n4) {
  int i = blockIdx.x * blockDim.x + threadIdx.x;
  int stride = gridDim.x * blockDim.x;
  for (; i < n4; i += stride) {
    float4 f = reinterpret_cast<const float4*>(src)[i];
    __align__(8) __bf16 o[4] = { (__bf16)f.x, (__bf16)f.y, (__bf16)f.z, (__bf16)f.w };
    *reinterpret_cast<uint2*>(dst + 4 * (size_t)i) = *reinterpret_cast<const uint2*>(o);
  }
}

// ---------------------------------------------------------------------------
// TN GEMM: C[M,N] = A[M,K] * Bm[N,K]^T (+ bias), bf16 in, f32 accum.
// Block = 256 threads = 8 waves. Block tile 128(M) x 128(N), K stepped by 32,
// double-buffered LDS ping-pong, one barrier per K step. Staging is async
// global->LDS (ASYNCcnt) when available, else global->VGPR->ds_store.
// Wave w computes rows [w*16,w*16+16) x 128 cols: 8 v8f accumulators, one A
// fragment reused 8x, B fragments software-pipelined in two register sets.
// Fragment layouts per CDNA5 ISA 7.12.2.
// ---------------------------------------------------------------------------
template<int WITH_BIAS, int OUT_BF16>
__global__ __launch_bounds__(256)
void gemm_tn_wmma(const __bf16* __restrict__ A,    // [M][K] row-major
                  const __bf16* __restrict__ Bm,   // [N][K] row-major
                  const float*  __restrict__ bias, // [N] or nullptr
                  void* __restrict__ Cout,         // [M][N] f32 or bf16
                  int M, int N, int K)
{
  __shared__ __align__(16) __bf16 lA[2][128 * 32];  // lA[p][r*32 + k]
  __shared__ __align__(16) __bf16 lB[2][128 * 32];  // lB[p][col*32 + k]

  const int t    = threadIdx.x;
  const int wave = t >> 5;
  const int lane = t & 31;
  const int half = lane >> 4;
  const int ln16 = lane & 15;

  const int m0 = blockIdx.y * 128;
  const int n0 = blockIdx.x * 128;

  v8f acc[8] = {};

  const uint4* Ag = reinterpret_cast<const uint4*>(A);
  const uint4* Bg = reinterpret_cast<const uint4*>(Bm);
  const int K8 = K >> 3;                 // row length in uint4 (8 bf16 each)

  // Each 128x32 bf16 tile = 512 uint4; thread t stages entries t and t+256.
  const int r0 = t >> 2,        c0 = t & 3;
  const int r1 = 64 + (t >> 2), c1 = t & 3;

#if USE_ASYNC_LDS
  // ---- async staging: L2 -> LDS directly, tracked on ASYNCcnt ----
  auto issue_tile = [&](int kk, int q) {
    const int ko = kk >> 3;
    lv4i_ptr dA = LPTR128(lA[q]);
    lv4i_ptr dB = LPTR128(lB[q]);
    __builtin_amdgcn_global_load_async_to_lds_b128(
        GPTR128(Ag + (size_t)(m0 + r0) * K8 + ko + c0), dA + t, 0, 0);
    __builtin_amdgcn_global_load_async_to_lds_b128(
        GPTR128(Ag + (size_t)(m0 + r1) * K8 + ko + c1), dA + t + 256, 0, 0);
    __builtin_amdgcn_global_load_async_to_lds_b128(
        GPTR128(Bg + (size_t)(n0 + r0) * K8 + ko + c0), dB + t, 0, 0);
    __builtin_amdgcn_global_load_async_to_lds_b128(
        GPTR128(Bg + (size_t)(n0 + r1) * K8 + ko + c1), dB + t + 256, 0, 0);
  };
  issue_tile(0, 0);
  __builtin_amdgcn_s_wait_asynccnt(0);
  __syncthreads();
#else
  // ---- register staging fallback ----
  uint4 pa0 = Ag[(size_t)(m0 + r0) * K8 + c0];
  uint4 pa1 = Ag[(size_t)(m0 + r1) * K8 + c1];
  uint4 pb0 = Bg[(size_t)(n0 + r0) * K8 + c0];
  uint4 pb1 = Bg[(size_t)(n0 + r1) * K8 + c1];
  reinterpret_cast<uint4*>(lA[0])[t]       = pa0;
  reinterpret_cast<uint4*>(lA[0])[t + 256] = pa1;
  reinterpret_cast<uint4*>(lB[0])[t]       = pb0;
  reinterpret_cast<uint4*>(lB[0])[t + 256] = pb1;
  __syncthreads();
#endif

  int p = 0;
  for (int kk = 0; kk < K; kk += 32) {
    const bool more = (kk + 32) < K;

#if USE_ASYNC_LDS
    if (more) issue_tile(kk + 32, p ^ 1);   // in flight during the WMMA burst
#else
    if (more) {
      const int ko = (kk + 32) >> 3;
      pa0 = Ag[(size_t)(m0 + r0) * K8 + ko + c0];
      pa1 = Ag[(size_t)(m0 + r1) * K8 + ko + c1];
      pb0 = Bg[(size_t)(n0 + r0) * K8 + ko + c0];
      pb1 = Bg[(size_t)(n0 + r1) * K8 + ko + c1];
    }
#endif

    // A fragment (16x32 bf16): VGPR v holds K = (v/4)*16 + half*8 + (v%4)*2
    union BF { v16bf v; unsigned u[8]; };
    BF afrag;
    #pragma unroll
    for (int v = 0; v < 8; ++v) {
      int kb = (v >> 2) * 16 + half * 8 + (v & 3) * 2;
      afrag.u[v] = *reinterpret_cast<const unsigned*>(
          &lA[p][(wave * 16 + ln16) * 32 + kb]);
    }

    // B fragments, software-pipelined across two register sets: the ds_loads
    // for subtile s+1 are issued before the WMMA for subtile s.
    BF bfr[2];
    #pragma unroll
    for (int v = 0; v < 8; ++v) {
      bfr[0].u[v] = *reinterpret_cast<const unsigned*>(
          &lB[p][ln16 * 32 + (v >> 2) * 16 + half * 8 + (v & 3) * 2]);
    }
    #pragma unroll
    for (int s = 0; s < 8; ++s) {
      if (s < 7) {
        int col = (s + 1) * 16 + ln16;
        #pragma unroll
        for (int v = 0; v < 8; ++v) {
          bfr[(s + 1) & 1].u[v] = *reinterpret_cast<const unsigned*>(
              &lB[p][col * 32 + (v >> 2) * 16 + half * 8 + (v & 3) * 2]);
        }
      }
      acc[s] = __builtin_amdgcn_wmma_f32_16x16x32_bf16(
          false, afrag.v, false, bfr[s & 1].v, (short)0, acc[s], false, false);
    }

    if (more) {
      const int q = p ^ 1;
#if USE_ASYNC_LDS
      __builtin_amdgcn_s_wait_asynccnt(0);
#else
      reinterpret_cast<uint4*>(lA[q])[t]       = pa0;
      reinterpret_cast<uint4*>(lA[q])[t + 256] = pa1;
      reinterpret_cast<uint4*>(lB[q])[t]       = pb0;
      reinterpret_cast<uint4*>(lB[q])[t + 256] = pb1;
#endif
      __syncthreads();
      p = q;
    }
  }

  // Epilogue: C/D layout -> element (M = v + 8*half, N = ln16) per VGPR v.
  #pragma unroll
  for (int s = 0; s < 8; ++s) {
    int n = n0 + s * 16 + ln16;
    float bv = WITH_BIAS ? bias[n] : 0.0f;
    #pragma unroll
    for (int v = 0; v < 8; ++v) {
      int m = m0 + wave * 16 + v + 8 * half;
      float val = acc[s][v] + bv;
      if (OUT_BF16) {
        reinterpret_cast<__bf16*>(Cout)[(size_t)m * N + n] = (__bf16)val;
      } else {
        reinterpret_cast<float*>(Cout)[(size_t)m * N + n] = val;
      }
    }
  }
}

// ---------------------------------------------------------------------------
// Row softmax, one block per row, 8192 = 256 threads x 32 register-resident
// values -> single read + single write of the 134 MB matrix.
// ---------------------------------------------------------------------------
__global__ __launch_bounds__(256)
void softmax_row_kernel(float* __restrict__ E) {
  float* p = E + (size_t)blockIdx.x * SEQ_LEN;
  const int t = threadIdx.x;
  float v[32];

  float mx = -3.402823e38f;
  #pragma unroll
  for (int j = 0; j < 32; ++j) {
    v[j] = p[t + j * 256];
    mx = fmaxf(mx, v[j]);
  }

  __shared__ float red[256];
  red[t] = mx;
  __syncthreads();
  for (int s = 128; s > 0; s >>= 1) {
    if (t < s) red[t] = fmaxf(red[t], red[t + s]);
    __syncthreads();
  }
  mx = red[0];
  __syncthreads();

  float sum = 0.0f;
  #pragma unroll
  for (int j = 0; j < 32; ++j) {
    v[j] = __expf(v[j] - mx);
    sum += v[j];
  }
  red[t] = sum;
  __syncthreads();
  for (int s = 128; s > 0; s >>= 1) {
    if (t < s) red[t] += red[t + s];
    __syncthreads();
  }
  float inv = 1.0f / red[0];

  #pragma unroll
  for (int j = 0; j < 32; ++j) p[t + j * 256] = v[j] * inv;
}

// ---------------------------------------------------------------------------
extern "C" void kernel_launch(void* const* d_in, const int* in_sizes, int n_in,
                              void* d_out, int out_size, void* d_ws, size_t ws_size,
                              hipStream_t stream) {
  (void)in_sizes; (void)n_in; (void)out_size; (void)ws_size;

  const float* out_state = (const float*)d_in[0];  // [4096,1024]
  const float* history   = (const float*)d_in[1];  // [8192,1024]
  const float* W         = (const float*)d_in[2];  // [1024,1024]
  const float* bias      = (const float*)d_in[3];  // [1024]

  // Workspace layout (bf16): history copy, out_state copy, W copy, proj.
  __bf16* hist_bf  = (__bf16*)d_ws;
  __bf16* state_bf = hist_bf  + (size_t)SEQ_LEN   * HIDDEN_DIM;
  __bf16* w_bf     = state_bf + (size_t)STATE_LEN * HIDDEN_DIM;
  __bf16* proj_bf  = w_bf     + (size_t)HIDDEN_DIM * HIDDEN_DIM;

  // 1) f32 -> bf16 converts (L2-resident afterwards).
  {
    int n4 = SEQ_LEN * HIDDEN_DIM / 4;
    f32_to_bf16_kernel<<<n4 / 256, 256, 0, stream>>>(history, hist_bf, n4);
    n4 = STATE_LEN * HIDDEN_DIM / 4;
    f32_to_bf16_kernel<<<n4 / 256, 256, 0, stream>>>(out_state, state_bf, n4);
    n4 = HIDDEN_DIM * HIDDEN_DIM / 4;
    f32_to_bf16_kernel<<<n4 / 256, 256, 0, stream>>>(W, w_bf, n4);
  }

  // 2) proj = history @ W^T + b  (bf16 out)  [8192,1024]
  gemm_tn_wmma<1, 1><<<dim3(HIDDEN_DIM / 128, SEQ_LEN / 128), 256, 0, stream>>>(
      hist_bf, w_bf, bias, proj_bf, SEQ_LEN, HIDDEN_DIM, HIDDEN_DIM);

  // 3) energies = out_state @ proj^T  (f32 out into d_out)  [4096,8192]
  gemm_tn_wmma<0, 0><<<dim3(SEQ_LEN / 128, STATE_LEN / 128), 256, 0, stream>>>(
      state_bf, proj_bf, nullptr, d_out, STATE_LEN, SEQ_LEN, HIDDEN_DIM);

  // 4) row softmax in place on d_out.
  softmax_row_kernel<<<STATE_LEN, 256, 0, stream>>>((float*)d_out);
}
